// Talkotron_67250597920863
// MI455X (gfx1250) — matrix operand order repaired and checked by reading, not compiled, not run on hardware
//
#include <hip/hip_runtime.h>

// ---------------------------------------------------------------------------
// Talkotron autoregressive decoder for MI455X (gfx1250, wave32, WMMA bf16).
// All dense matmuls use v_wmma_f32_16x16x32_bf16; weights converted to padded
// bf16 once per launch. 4 kernels per decode step, stream-ordered, 250 steps.
// ---------------------------------------------------------------------------

typedef __attribute__((ext_vector_type(16))) __bf16 v16bf;
typedef __attribute__((ext_vector_type(8)))  __bf16 v8bf;
typedef __attribute__((ext_vector_type(8)))  float  v8f;

#define DEV __device__ __forceinline__

DEV v8f vbcast(float v){
  v8f r;
  #pragma unroll
  for(int i=0;i<8;++i) r[i]=v;
  return r;
}

// A-operand (16x32 bf16 tile): lane<16 holds row (lane), K {0..7,16..23};
// lane>=16 same row, K {8..15,24..31}.  p0 = tile origin (row0, k0).
DEV v16bf load_a(const __bf16* p0, int ld, int lane){
  const __bf16* p = p0 + (lane&15)*ld + ((lane&16)?8:0);
  v8bf lo = *(const v8bf*)p;
  v8bf hi = *(const v8bf*)(p+16);
  v16bf r;
  #pragma unroll
  for(int i=0;i<8;++i){ r[i]=lo[i]; r[8+i]=hi[i]; }
  return r;
}

// B-operand (32x16): lane n holds output-feature row n of W[out,in];
// lane<16 -> K {0..7,16..23}, lane>=16 -> K {8..15,24..31}.
DEV v16bf load_b(const __bf16* p0, int ld, int lane){
  const __bf16* p = p0 + (lane&15)*ld + ((lane&16)?8:0);
  v8bf lo = *(const v8bf*)p;
  v8bf hi = *(const v8bf*)(p+16);
  v16bf r;
  #pragma unroll
  for(int i=0;i<8;++i){ r[i]=lo[i]; r[8+i]=hi[i]; }
  return r;
}

DEV v8f wmma_bf(v16bf a, v16bf b, v8f c){
  return __builtin_amdgcn_wmma_f32_16x16x32_bf16(false,a,false,b,(short)0,c,false,false);
}

DEV v8f gemm_acc(const __bf16* A,int lda,const __bf16* Bw,int ldb,int K,int lane,v8f acc){
  for(int k=0;k<K;k+=32)
    acc = wmma_bf(load_a(A+k,lda,lane), load_b(Bw+k,ldb,lane), acc);
  return acc;
}

// three GEMMs sharing the A operand (GRU r/z/n gates)
DEV void tri_gemm(v8f&c0,v8f&c1,v8f&c2,const __bf16*A,int lda,
                  const __bf16*B0,const __bf16*B1,const __bf16*B2,int ldb,int K,int lane){
  for(int k=0;k<K;k+=32){
    v16bf a=load_a(A+k,lda,lane);
    c0=wmma_bf(a,load_b(B0+k,ldb,lane),c0);
    c1=wmma_bf(a,load_b(B1+k,ldb,lane),c1);
    c2=wmma_bf(a,load_b(B2+k,ldb,lane),c2);
  }
}

DEV float sigmf(float x){ return 1.f/(1.f+__expf(-x)); }

// ---------------------------------------------------------------------------
// Prep kernels
// ---------------------------------------------------------------------------
__global__ __launch_bounds__(256) void k_cvt(const float* __restrict__ src, __bf16* __restrict__ dst,
                                             int srows,int scols,int drows,int dcols){
  int n=drows*dcols;
  for(int i=blockIdx.x*blockDim.x+threadIdx.x;i<n;i+=gridDim.x*blockDim.x){
    int r=i/dcols,c=i%dcols;
    float v=(r<srows&&c<scols)?src[r*scols+c]:0.f;
    dst[i]=(__bf16)v;
  }
}

__global__ __launch_bounds__(256) void k_build_wc(const float* __restrict__ dU,const float* __restrict__ dT,
                                                  __bf16* __restrict__ wc){
  int i=blockIdx.x*blockDim.x+threadIdx.x;
  if(i>=128*32) return;
  int r=i/32,c=i%32;
  float v=(c<8)?dU[r*8+c]:(c<16?dT[r*8+(c-8)]:0.f);
  wc[i]=(__bf16)v;
}

__global__ void k_prior(float* prior){
  int k=threadIdx.x;
  if(k>10) return;
  const float n=10.f,a=0.1f,b=0.9f;
  float kk=(float)k;
  float logC =lgammaf(n+1.f)-lgammaf(kk+1.f)-lgammaf(n-kk+1.f);
  float logB =lgammaf(kk+a)+lgammaf(n-kk+b)-lgammaf(n+a+b);
  float logB0=lgammaf(a)+lgammaf(b)-lgammaf(a+b);
  prior[10-k]=__expf(logC+logB-logB0);   // flipped, as in reference
}

__global__ __launch_bounds__(256) void k_zero(char* p,size_t n){
  size_t i=(size_t)blockIdx.x*blockDim.x+threadIdx.x;
  size_t st=(size_t)gridDim.x*blockDim.x;
  for(;i<n;i+=st) p[i]=0;
}

__global__ void k_alpha_init(float* alpha){
  int b=threadIdx.x;
  if(b<64) alpha[b*512]=1.f;
}

// ---------------------------------------------------------------------------
// Kernel A: prenet + attention GRU + dynamic-filter MLP  (grid=4, 16 rows/blk)
// ---------------------------------------------------------------------------
__global__ __launch_bounds__(512) void k_att_rnn(
    const __bf16* __restrict__ frameb,    // [64][416]
    const __bf16* __restrict__ ctxb,      // [64][768]
    const float*  __restrict__ hattf_old, float* __restrict__ hattf_new,
    const __bf16* __restrict__ hattb_old, __bf16* __restrict__ hattb_new,
    const __bf16* __restrict__ w_p1, const float* __restrict__ b_p1,
    const __bf16* __restrict__ w_p2, const float* __restrict__ b_p2,
    const __bf16* __restrict__ w_awi, const float* __restrict__ b_ai,
    const __bf16* __restrict__ w_awh, const float* __restrict__ b_ah,
    const __bf16* __restrict__ w_dW,  const float* __restrict__ b_dW,
    const __bf16* __restrict__ w_dV,
    float* __restrict__ Gout)             // [64][176]
{
  __shared__ __bf16 sP1[16][256];   // pre1, later new h_att (bf16)
  __shared__ __bf16 sP2[16][128];   // pre2
  __shared__ __bf16 sT [16][128];   // tanh(h @ dW^T)
  const int tid=threadIdx.x, wave=tid>>5, lane=tid&31;
  const int hi8=(lane&16)?8:0, ln=lane&15;
  const int m0=blockIdx.x*16;

  // phase 1: pre1 = relu(frame @ p1^T + b)   [16 x 256], K=416
  for(int nt=wave;nt<16;nt+=16){
    int col=nt*16+ln;
    v8f acc=vbcast(b_p1[col]);
    acc=gemm_acc(frameb+m0*416,416, w_p1+nt*16*416,416, 416,lane,acc);
    #pragma unroll
    for(int i=0;i<8;++i) sP1[i+hi8][col]=(__bf16)fmaxf(acc[i],0.f);
  }
  __syncthreads();

  // phase 2: pre2 = relu(pre1 @ p2^T + b)    [16 x 128], K=256
  for(int nt=wave;nt<8;nt+=16){
    int col=nt*16+ln;
    v8f acc=vbcast(b_p2[col]);
    acc=gemm_acc(&sP1[0][0],256, w_p2+nt*16*256,256, 256,lane,acc);
    #pragma unroll
    for(int i=0;i<8;++i) sP2[i+hi8][col]=(__bf16)fmaxf(acc[i],0.f);
  }
  __syncthreads();

  // phase 3: attention GRUCell, input=[ctx(768), pre2(128)], hidden 256
  for(int ct=wave;ct<16;ct+=16){
    int col=ct*16+ln;
    v8f ir=vbcast(b_ai[col]), iz=vbcast(b_ai[256+col]), in_=vbcast(b_ai[512+col]);
    v8f hr=vbcast(b_ah[col]), hz=vbcast(b_ah[256+col]), hn=vbcast(b_ah[512+col]);
    const __bf16 *Wr=w_awi+(ct*16)*896, *Wz=w_awi+(256+ct*16)*896, *Wn=w_awi+(512+ct*16)*896;
    tri_gemm(ir,iz,in_, ctxb+m0*768,768, Wr,Wz,Wn, 896, 768, lane);          // ctx part
    tri_gemm(ir,iz,in_, &sP2[0][0],128, Wr+768,Wz+768,Wn+768, 896, 128, lane); // pre part
    const __bf16 *Vr=w_awh+(ct*16)*256, *Vz=w_awh+(256+ct*16)*256, *Vn=w_awh+(512+ct*16)*256;
    tri_gemm(hr,hz,hn, hattb_old+m0*256,256, Vr,Vz,Vn, 256, 256, lane);
    #pragma unroll
    for(int i=0;i<8;++i){
      int row=m0+i+hi8;
      float hold=hattf_old[row*256+col];
      float r=sigmf(ir[i]+hr[i]);
      float z=sigmf(iz[i]+hz[i]);
      float nn=tanhf(in_[i]+r*hn[i]);
      float h=(1.f-z)*nn+z*hold;
      hattf_new[row*256+col]=h;
      hattb_new[row*256+col]=(__bf16)h;
      sP1[i+hi8][col]=(__bf16)h;
    }
  }
  __syncthreads();

  // phase 4: sT = tanh(h' @ dW^T + b)        [16 x 128], K=256
  for(int nt=wave;nt<8;nt+=16){
    int col=nt*16+ln;
    v8f acc=vbcast(b_dW[col]);
    acc=gemm_acc(&sP1[0][0],256, w_dW+nt*16*256,256, 256,lane,acc);
    #pragma unroll
    for(int i=0;i<8;++i) sT[i+hi8][col]=(__bf16)tanhf(acc[i]);
  }
  __syncthreads();

  // phase 5: G = sT @ dV^T                   [16 x 176], K=128
  for(int nt=wave;nt<11;nt+=16){
    int col=nt*16+ln;
    v8f acc=vbcast(0.f);
    acc=gemm_acc(&sT[0][0],128, w_dV+nt*16*128,128, 128,lane,acc);
    #pragma unroll
    for(int i=0;i<8;++i) Gout[(m0+i+hi8)*176+col]=acc[i];
  }
}

// ---------------------------------------------------------------------------
// Kernel B: DCA attention -> new alpha.  One workgroup per batch row.
// ---------------------------------------------------------------------------
__global__ __launch_bounds__(256) void k_dca(
    float* __restrict__ alpha,        // [64][512] in/out
    const float* __restrict__ G,      // [64][176]
    const float* __restrict__ dF,     // [8*21]
    const __bf16* __restrict__ wc,    // [128][32] = [dU | dT | 0]
    const float* __restrict__ dT_b,   // [128]
    const float* __restrict__ dv,     // [128]
    const float* __restrict__ prior)  // [11]
{
  __shared__ float  sAp[544];         // zero-padded alpha (10 each side)
  __shared__ __bf16 sFG[512][32];     // [f(8) | g(8) | 0(16)]
  __shared__ float  sE[512];
  __shared__ float  sred[256];
  __shared__ float  sGk[168], sFk[168], sPr[11];
  const int b=blockIdx.x, tid=threadIdx.x, wave=tid>>5, lane=tid&31;
  const int hi8=(lane&16)?8:0, ln=lane&15;

  for(int i=tid;i<544;i+=256) sAp[i]=(i>=10&&i<522)?alpha[b*512+i-10]:0.f;
  for(int i=tid;i<168;i+=256){ sGk[i]=G[b*176+i]; sFk[i]=dF[i]; }
  if(tid<11) sPr[tid]=prior[tid];
  __syncthreads();

  // prior (log) + static/dynamic convs
  for(int s=tid;s<512;s+=256){
    float p=0.f;
    #pragma unroll
    for(int j=0;j<11;++j) p=fmaf(sAp[s+j],sPr[j],p);
    sE[s]=logf(fmaxf(p,1e-6f));
    #pragma unroll
    for(int c=0;c<8;++c){
      float f=0.f,g=0.f;
      #pragma unroll
      for(int k=0;k<21;++k){
        float a=sAp[s+k];
        f=fmaf(a,sFk[c*21+k],f);
        g=fmaf(a,sGk[c*21+k],g);
      }
      sFG[s][c]=(__bf16)f; sFG[s][8+c]=(__bf16)g;
    }
    #pragma unroll
    for(int c=16;c<32;++c) sFG[s][c]=(__bf16)0.f;
  }
  __syncthreads();

  // e += tanh([f,g] @ [dU|dT]^T + b) . dv   via WMMA + LDS atomic reduction
  for(int job=wave;job<256;job+=8){
    int mt=job>>3, nt=job&7, col=nt*16+ln;
    v8f acc=vbcast(dT_b[col]);
    acc=gemm_acc(&sFG[mt*16][0],32, wc+nt*16*32,32, 32,lane,acc);
    float dvc=dv[col];
    #pragma unroll
    for(int i=0;i<8;++i) atomicAdd(&sE[mt*16+i+hi8], tanhf(acc[i])*dvc);
  }
  __syncthreads();

  // softmax over S=512
  float e0=sE[tid], e1=sE[tid+256];
  sred[tid]=fmaxf(e0,e1); __syncthreads();
  for(int o=128;o>0;o>>=1){ if(tid<o) sred[tid]=fmaxf(sred[tid],sred[tid+o]); __syncthreads(); }
  float mx=sred[0]; __syncthreads();
  float x0=__expf(e0-mx), x1=__expf(e1-mx);
  sred[tid]=x0+x1; __syncthreads();
  for(int o=128;o>0;o>>=1){ if(tid<o) sred[tid]+=sred[tid+o]; __syncthreads(); }
  float inv=1.f/sred[0];
  alpha[b*512+tid]=x0*inv;
  alpha[b*512+tid+256]=x1*inv;
}

// ---------------------------------------------------------------------------
// Kernel C: ctx = alpha @ enc (per-batch matvec, enc stays L2-resident)
// ---------------------------------------------------------------------------
__global__ __launch_bounds__(256) void k_ctx(const float* __restrict__ alpha,
                                             const float* __restrict__ enc,
                                             __bf16* __restrict__ ctxb){
  const int b=blockIdx.y;
  const int col=blockIdx.x*256+threadIdx.x;       // 0..767
  __shared__ float sAl[512];
  for(int i=threadIdx.x;i<512;i+=256) sAl[i]=alpha[b*512+i];
  __syncthreads();
  const float* e=enc+(size_t)b*512*768+col;
  float acc=0.f;
  #pragma unroll 4
  for(int s=0;s<512;++s) acc=fmaf(sAl[s], e[(size_t)s*768], acc);
  ctxb[b*768+col]=(__bf16)acc;
}

// ---------------------------------------------------------------------------
// Kernel D: linear + GRU1 + GRU2 (residual) + projection; scatter into d_out
// ---------------------------------------------------------------------------
__global__ __launch_bounds__(512) void k_dec_rnn(
    const __bf16* __restrict__ ctxb, const __bf16* __restrict__ hattb,
    const __bf16* __restrict__ w_l,  const float* __restrict__ b_l,
    const __bf16* __restrict__ w_g1i,const float* __restrict__ b_g1i,
    const __bf16* __restrict__ w_g1h,const float* __restrict__ b_g1h,
    const float* __restrict__ h1f_old, float* __restrict__ h1f_new,
    const __bf16* __restrict__ h1b_old,__bf16* __restrict__ h1b_new,
    const __bf16* __restrict__ w_g2i,const float* __restrict__ b_g2i,
    const __bf16* __restrict__ w_g2h,const float* __restrict__ b_g2h,
    const float* __restrict__ h2f_old, float* __restrict__ h2f_new,
    const __bf16* __restrict__ h2b_old,__bf16* __restrict__ h2b_new,
    const __bf16* __restrict__ w_pr, const float* __restrict__ b_pr,
    float* __restrict__ out, __bf16* __restrict__ frameb, int t)
{
  __shared__ __bf16 sX [16][256];   // x bf16, then x3 bf16
  __shared__ __bf16 sX2[16][256];   // x2 bf16
  __shared__ float  sXf[16][256];   // x fp32, then x2 fp32
  const int tid=threadIdx.x, wave=tid>>5, lane=tid&31;
  const int hi8=(lane&16)?8:0, ln=lane&15;
  const int m0=blockIdx.x*16;

  // phase 1: x = [ctx, h_att] @ l^T + b      [16 x 256], K=1024
  for(int nt=wave;nt<16;nt+=16){
    int col=nt*16+ln;
    v8f acc=vbcast(b_l[col]);
    const __bf16* Bb=w_l+nt*16*1024;
    acc=gemm_acc(ctxb+m0*768,768, Bb,1024, 768,lane,acc);
    acc=gemm_acc(hattb+m0*256,256, Bb+768,1024, 256,lane,acc);
    #pragma unroll
    for(int i=0;i<8;++i){ sX[i+hi8][col]=(__bf16)acc[i]; sXf[i+hi8][col]=acc[i]; }
  }
  __syncthreads();

  // phase 2: GRU1 + residual -> x2
  for(int ct=wave;ct<16;ct+=16){
    int col=ct*16+ln;
    v8f ir=vbcast(b_g1i[col]), iz=vbcast(b_g1i[256+col]), in_=vbcast(b_g1i[512+col]);
    v8f hr=vbcast(b_g1h[col]), hz=vbcast(b_g1h[256+col]), hn=vbcast(b_g1h[512+col]);
    tri_gemm(ir,iz,in_, &sX[0][0],256,
             w_g1i+(ct*16)*256, w_g1i+(256+ct*16)*256, w_g1i+(512+ct*16)*256, 256, 256, lane);
    tri_gemm(hr,hz,hn, h1b_old+m0*256,256,
             w_g1h+(ct*16)*256, w_g1h+(256+ct*16)*256, w_g1h+(512+ct*16)*256, 256, 256, lane);
    #pragma unroll
    for(int i=0;i<8;++i){
      int row=m0+i+hi8;
      float hold=h1f_old[row*256+col];
      float r=sigmf(ir[i]+hr[i]);
      float z=sigmf(iz[i]+hz[i]);
      float nn=tanhf(in_[i]+r*hn[i]);
      float h=(1.f-z)*nn+z*hold;
      h1f_new[row*256+col]=h; h1b_new[row*256+col]=(__bf16)h;
      float x2=sXf[i+hi8][col]+h;
      sXf[i+hi8][col]=x2; sX2[i+hi8][col]=(__bf16)x2;
    }
  }
  __syncthreads();

  // phase 3: GRU2 + residual -> x3 (into sX)
  for(int ct=wave;ct<16;ct+=16){
    int col=ct*16+ln;
    v8f ir=vbcast(b_g2i[col]), iz=vbcast(b_g2i[256+col]), in_=vbcast(b_g2i[512+col]);
    v8f hr=vbcast(b_g2h[col]), hz=vbcast(b_g2h[256+col]), hn=vbcast(b_g2h[512+col]);
    tri_gemm(ir,iz,in_, &sX2[0][0],256,
             w_g2i+(ct*16)*256, w_g2i+(256+ct*16)*256, w_g2i+(512+ct*16)*256, 256, 256, lane);
    tri_gemm(hr,hz,hn, h2b_old+m0*256,256,
             w_g2h+(ct*16)*256, w_g2h+(256+ct*16)*256, w_g2h+(512+ct*16)*256, 256, 256, lane);
    #pragma unroll
    for(int i=0;i<8;++i){
      int row=m0+i+hi8;
      float hold=h2f_old[row*256+col];
      float r=sigmf(ir[i]+hr[i]);
      float z=sigmf(iz[i]+hz[i]);
      float nn=tanhf(in_[i]+r*hn[i]);
      float h=(1.f-z)*nn+z*hold;
      h2f_new[row*256+col]=h; h2b_new[row*256+col]=(__bf16)h;
      sX[i+hi8][col]=(__bf16)(sXf[i+hi8][col]+h);
    }
  }
  __syncthreads();

  // phase 4: out = x3 @ pr^T + b [16 x 400]; scatter to [B,80,T*5] + next frame
  for(int nt=wave;nt<25;nt+=16){
    int col=nt*16+ln;                               // 0..399
    v8f acc=vbcast(b_pr[col]);
    acc=gemm_acc(&sX[0][0],256, w_pr+nt*16*256,256, 256,lane,acc);
    int mm=col/5, rr=col%5;
    #pragma unroll
    for(int i=0;i<8;++i){
      int row=m0+i+hi8;                             // batch index
      float v=acc[i];
      frameb[row*416+col]=(__bf16)v;
      out[row*100000 + mm*1250 + t*5 + rr]=v;
    }
  }
}

// ---------------------------------------------------------------------------
extern "C" void kernel_launch(void* const* d_in, const int* in_sizes, int n_in,
                              void* d_out, int out_size, void* d_ws, size_t ws_size,
                              hipStream_t stream) {
  (void)in_sizes;(void)n_in;(void)out_size;(void)ws_size;
  const float* enc  =(const float*)d_in[0];
  const float* p1_w =(const float*)d_in[1];  const float* p1_b =(const float*)d_in[2];
  const float* p2_w =(const float*)d_in[3];  const float* p2_b =(const float*)d_in[4];
  const float* a_wi =(const float*)d_in[5];  const float* a_bi =(const float*)d_in[6];
  const float* a_wh =(const float*)d_in[7];  const float* a_bh =(const float*)d_in[8];
  const float* dW_w =(const float*)d_in[9];  const float* dW_b =(const float*)d_in[10];
  const float* dV_w =(const float*)d_in[11];
  const float* dF_w =(const float*)d_in[12];
  const float* dU_w =(const float*)d_in[13];
  const float* dT_w =(const float*)d_in[14]; const float* dT_b =(const float*)d_in[15];
  const float* dv_w =(const float*)d_in[16];
  const float* l_w  =(const float*)d_in[17]; const float* l_b  =(const float*)d_in[18];
  const float* g1_wi=(const float*)d_in[19]; const float* g1_bi=(const float*)d_in[20];
  const float* g1_wh=(const float*)d_in[21]; const float* g1_bh=(const float*)d_in[22];
  const float* g2_wi=(const float*)d_in[23]; const float* g2_bi=(const float*)d_in[24];
  const float* g2_wh=(const float*)d_in[25]; const float* g2_bh=(const float*)d_in[26];
  const float* pr_w =(const float*)d_in[27]; const float* pr_b =(const float*)d_in[28];
  const int T = 250; // == setup_inputs()['T_dec'] (device scalar not readable during capture)

  char* ws=(char*)d_ws; size_t off=0;
  auto alloc=[&](size_t bytes)->char*{ off=(off+255)&~(size_t)255; char* p=ws+off; off+=bytes; return p; };

  // bf16 weights (padded: K -> mult of 32, N -> mult of 16)
  __bf16* wb_p1 =(__bf16*)alloc(256*416*2);
  __bf16* wb_p2 =(__bf16*)alloc(128*256*2);
  __bf16* wb_awi=(__bf16*)alloc(768*896*2);
  __bf16* wb_awh=(__bf16*)alloc(768*256*2);
  __bf16* wb_dW =(__bf16*)alloc(128*256*2);
  __bf16* wb_dV =(__bf16*)alloc(176*128*2);
  __bf16* wb_wc =(__bf16*)alloc(128*32*2);
  __bf16* wb_l  =(__bf16*)alloc(256*1024*2);
  __bf16* wb_g1i=(__bf16*)alloc(768*256*2);
  __bf16* wb_g1h=(__bf16*)alloc(768*256*2);
  __bf16* wb_g2i=(__bf16*)alloc(768*256*2);
  __bf16* wb_g2h=(__bf16*)alloc(768*256*2);
  __bf16* wb_pr =(__bf16*)alloc(400*256*2);
  float*  prior =(float*)alloc(11*4);

  // recurrent state (zeroed below)
  char* state0=ws+((off+255)&~(size_t)255);
  __bf16* frameb=(__bf16*)alloc(64*416*2);
  __bf16* ctxb  =(__bf16*)alloc(64*768*2);
  float*  hattf[2]; __bf16* hattb[2];
  float*  h1f[2];   __bf16* h1b[2];
  float*  h2f[2];   __bf16* h2b[2];
  for(int i=0;i<2;++i){ hattf[i]=(float*)alloc(64*256*4); hattb[i]=(__bf16*)alloc(64*256*2); }
  for(int i=0;i<2;++i){ h1f[i]=(float*)alloc(64*256*4); h1b[i]=(__bf16*)alloc(64*256*2); }
  for(int i=0;i<2;++i){ h2f[i]=(float*)alloc(64*256*4); h2b[i]=(__bf16*)alloc(64*256*2); }
  float* alpha=(float*)alloc(64*512*4);
  float* Gbuf =(float*)alloc(64*176*4);
  size_t state_bytes=(size_t)((ws+off)-state0);

  // ---- prep ----
  auto cvt=[&](const float* s,__bf16* d,int sr,int sc,int dr,int dc){
    int n=dr*dc; int blocks=(n+255)/256; if(blocks>2048) blocks=2048;
    k_cvt<<<blocks,256,0,stream>>>(s,d,sr,sc,dr,dc);
  };
  cvt(p1_w ,wb_p1 ,256,400,256,416);
  cvt(p2_w ,wb_p2 ,128,256,128,256);
  cvt(a_wi ,wb_awi,768,896,768,896);
  cvt(a_wh ,wb_awh,768,256,768,256);
  cvt(dW_w ,wb_dW ,128,256,128,256);
  cvt(dV_w ,wb_dV ,168,128,176,128);
  cvt(l_w  ,wb_l  ,256,1024,256,1024);
  cvt(g1_wi,wb_g1i,768,256,768,256);
  cvt(g1_wh,wb_g1h,768,256,768,256);
  cvt(g2_wi,wb_g2i,768,256,768,256);
  cvt(g2_wh,wb_g2h,768,256,768,256);
  cvt(pr_w ,wb_pr ,400,256,400,256);
  k_build_wc<<<(128*32+255)/256,256,0,stream>>>(dU_w,dT_w,wb_wc);
  k_prior<<<1,32,0,stream>>>(prior);
  {
    int blocks=(int)((state_bytes+255)/256); if(blocks>2048) blocks=2048;
    k_zero<<<blocks,256,0,stream>>>(state0,state_bytes);
  }
  k_alpha_init<<<1,64,0,stream>>>(alpha);

  // ---- decode loop ----
  float* out=(float*)d_out;
  for(int t=0;t<T;++t){
    int p=t&1, q=p^1;
    k_att_rnn<<<4,512,0,stream>>>(frameb,ctxb,
        hattf[p],hattf[q],hattb[p],hattb[q],
        wb_p1,p1_b, wb_p2,p2_b, wb_awi,a_bi, wb_awh,a_bh,
        wb_dW,dW_b, wb_dV, Gbuf);
    k_dca<<<64,256,0,stream>>>(alpha,Gbuf,dF_w,wb_wc,dT_b,dv_w,prior);
    k_ctx<<<dim3(3,64),256,0,stream>>>(alpha,enc,ctxb);
    k_dec_rnn<<<4,512,0,stream>>>(ctxb,hattb[q],
        wb_l,l_b,
        wb_g1i,g1_bi, wb_g1h,g1_bh, h1f[p],h1f[q],h1b[p],h1b[q],
        wb_g2i,g2_bi, wb_g2h,g2_bh, h2f[p],h2f[q],h2b[p],h2b[q],
        wb_pr,pr_b, out, frameb, t);
  }
}
